// BasicGCN_16329465660178
// MI455X (gfx1250) — compile-verified, matrix operand order checked by reading, not run on hardware
//
#include <hip/hip_runtime.h>
#include <hip/hip_bf16.h>

// ---------------------------------------------------------------------------
// BasicGCN forward for MI455X (gfx1250, wave32).
//  - Per-layer dense GEMM h@W runs on the WMMA pipe (f16 in, f32 accum).
//  - Message passing (gather/scale/scatter) uses f32 atomics; the whole
//    51 MB feature matrix fits in the 192 MB L2 so random row traffic is
//    L2-resident rather than HBM-bound.
// ---------------------------------------------------------------------------

typedef __attribute__((ext_vector_type(16))) _Float16 v16h;
typedef __attribute__((ext_vector_type(8)))  float    v8f;

#define HDIM 128
#define LDSK 136   // padded LDS stride (halves) for transposed W: avoids bank conflicts

// ---------------------------------------------------------------------------
// Utility kernels
// ---------------------------------------------------------------------------
__global__ void k_zero(float* __restrict__ p, long n) {
    long i = (long)blockIdx.x * blockDim.x + threadIdx.x;
    if (i < n) p[i] = 0.0f;
}

__global__ void k_deg(const int* __restrict__ dst, float* __restrict__ deg, int E) {
    int e = blockIdx.x * blockDim.x + threadIdx.x;
    if (e < E) atomicAdd(&deg[dst[e]], 1.0f);
}

__global__ void k_rsqrt(float* __restrict__ dinv, int N) {
    int i = blockIdx.x * blockDim.x + threadIdx.x;
    if (i < N) dinv[i] = rsqrtf(dinv[i] + 1.0f);   // +1 self-loop
}

// ---------------------------------------------------------------------------
// WMMA GEMM: Out[N x 128] = A[N x 128] @ W[128 x 128]  (f16 mul, f32 accum)
// Block: 256 threads (8 waves); 128 rows/block; each wave: 16 rows x 128 cols
// = 8 tiles of 16x16, K=128 -> 4 x v_wmma_f32_16x16x32_f16 per tile.
// Epilogue: wave-uniform full-tile test -> scalar branch, unguarded stores
// on the fast path (only the last of ~782 blocks ever takes the slow path).
// ---------------------------------------------------------------------------
__global__ __launch_bounds__(256) void k_gemm_wmma(
    const float* __restrict__ A, const float* __restrict__ W,
    float* __restrict__ Out, int Nrows)
{
    __shared__ _Float16 Wt[HDIM * LDSK];   // W transposed: Wt[n][k] = W[k][n]

    // stage W -> LDS as f16, transposed (coalesced global reads)
    for (int i = threadIdx.x; i < HDIM * HDIM; i += 256) {
        int k = i >> 7, n = i & 127;
        Wt[n * LDSK + k] = (_Float16)W[i];
    }
    __syncthreads();

    const int lane  = threadIdx.x & 31;
    const int wave  = threadIdx.x >> 5;
    const int mBase = blockIdx.x * 128 + wave * 16;
    const int mRow  = mBase + (lane & 15);
    const int mLd   = mRow < Nrows ? mRow : (Nrows - 1);  // clamp: keep EXEC full
    const int kHalf = (lane >> 4) << 3;                    // 0 or 8 (ISA A layout)
    const int hi8   = (lane >> 4) << 3;                    // row offset for C/D store
    const bool fullTile = (mBase + 16) <= Nrows;           // wave-uniform

    // A fragments for all 4 K-steps (ISA 16-bit A layout: K..K+7, K+16..K+23)
    v16h afrag[4];
    const float* __restrict__ arow = A + (size_t)mLd * HDIM;
    #pragma unroll
    for (int ks = 0; ks < 4; ++ks) {
        const int kb = ks * 32 + kHalf;
        #pragma unroll
        for (int i = 0; i < 8; ++i) {
            afrag[ks][i]     = (_Float16)arow[kb + i];
            afrag[ks][i + 8] = (_Float16)arow[kb + 16 + i];
        }
    }

    #pragma unroll
    for (int ct = 0; ct < 8; ++ct) {
        const int nCol = ct * 16 + (lane & 15);
        v8f c = {};
        #pragma unroll
        for (int ks = 0; ks < 4; ++ks) {
            const int kb = ks * 32 + kHalf;
            const _Float16* __restrict__ brow = &Wt[nCol * LDSK + kb];
            v16h b;
            #pragma unroll
            for (int i = 0; i < 8; ++i) { b[i] = brow[i]; b[i + 8] = brow[16 + i]; }
            c = __builtin_amdgcn_wmma_f32_16x16x32_f16(
                    false, afrag[ks], false, b, (short)0, c, false, false);
        }
        // D layout: VGPR v -> row mBase + v (+8 for upper lane half), col = lane&15
        float* __restrict__ orow = Out + (size_t)(mBase + hi8) * HDIM + nCol;
        if (fullTile) {
            #pragma unroll
            for (int v = 0; v < 8; ++v)
                orow[(size_t)v * HDIM] = c[v];
        } else {
            #pragma unroll
            for (int v = 0; v < 8; ++v) {
                if (mBase + hi8 + v < Nrows) orow[(size_t)v * HDIM] = c[v];
            }
        }
    }
}

// ---------------------------------------------------------------------------
// Self-loop init:  hout[i][:] = dinv[i]^2 * hin[i][:]   (also zero-initializes)
// ---------------------------------------------------------------------------
__global__ void k_self(const float* __restrict__ hin, const float* __restrict__ dinv,
                       float* __restrict__ hout, int N) {
    long idx = (long)blockIdx.x * blockDim.x + threadIdx.x;
    if (idx >= (long)N * HDIM) return;
    int n = (int)(idx >> 7);
    float di = dinv[n];
    hout[idx] = di * di * hin[idx];
}

// ---------------------------------------------------------------------------
// Edge scatter: one wave per edge, float4 per lane (32*4 = 128 features).
// s,d,w are wave-uniform -> scalar loads; row gather is one 512B L2 hit.
// ---------------------------------------------------------------------------
__global__ __launch_bounds__(256) void k_edge(
    const float* __restrict__ hin, const int* __restrict__ src,
    const int* __restrict__ dst, const float* __restrict__ dinv,
    float* __restrict__ hout, int E)
{
    int e = blockIdx.x * 8 + (threadIdx.x >> 5);
    if (e >= E) return;
    int lane = threadIdx.x & 31;
    int s = src[e], d = dst[e];
    float w = dinv[s] * dinv[d];
    float4 v = ((const float4*)(hin + (size_t)s * HDIM))[lane];
    float* o = hout + (size_t)d * HDIM + lane * 4;
    atomicAdd(o + 0, w * v.x);
    atomicAdd(o + 1, w * v.y);
    atomicAdd(o + 2, w * v.z);
    atomicAdd(o + 3, w * v.w);
}

// ---------------------------------------------------------------------------
// bias + BatchNorm(eval) + ReLU, in place
// ---------------------------------------------------------------------------
__global__ void k_bnrelu(float* __restrict__ h, const float* __restrict__ bias,
                         const float* __restrict__ gamma, const float* __restrict__ beta,
                         const float* __restrict__ mean, const float* __restrict__ var,
                         int N)
{
    long idx = (long)blockIdx.x * blockDim.x + threadIdx.x;
    if (idx >= (long)N * HDIM) return;
    int f = (int)(idx & 127);
    float v = h[idx] + bias[f];
    v = (v - mean[f]) * rsqrtf(var[f] + 1e-5f) * gamma[f] + beta[f];
    h[idx] = v > 0.0f ? v : 0.0f;
}

// ---------------------------------------------------------------------------
// Pooling: sum + max + count per graph. Post-ReLU values are >= 0 and the
// max buffer is zero-initialized, so integer-bit atomicMax == float max.
// ---------------------------------------------------------------------------
__global__ __launch_bounds__(256) void k_pool(
    const float* __restrict__ h, const int* __restrict__ batch,
    float* __restrict__ psum, float* __restrict__ pmax, float* __restrict__ pcnt,
    int N)
{
    int node = blockIdx.x * 8 + (threadIdx.x >> 5);
    if (node >= N) return;
    int lane = threadIdx.x & 31;
    int b = batch[node];
    if (lane == 0) atomicAdd(&pcnt[b], 1.0f);
    float4 v = ((const float4*)(h + (size_t)node * HDIM))[lane];
    float* ps = psum + (size_t)b * HDIM + lane * 4;
    float* pm = pmax + (size_t)b * HDIM + lane * 4;
    atomicAdd(ps + 0, v.x); atomicAdd(ps + 1, v.y);
    atomicAdd(ps + 2, v.z); atomicAdd(ps + 3, v.w);
    atomicMax((int*)(pm + 0), __float_as_int(v.x));
    atomicMax((int*)(pm + 1), __float_as_int(v.y));
    atomicMax((int*)(pm + 2), __float_as_int(v.z));
    atomicMax((int*)(pm + 3), __float_as_int(v.w));
}

// ---------------------------------------------------------------------------
// MLP head: one 128-thread block per graph. g=[mean|max|sum] (384) in LDS,
// fc1(384->128)+ReLU, fc2(128->64)+ReLU, fc3(64->1).
// ---------------------------------------------------------------------------
__global__ __launch_bounds__(128) void k_mlp(
    const float* __restrict__ psum, const float* __restrict__ pmax,
    const float* __restrict__ pcnt,
    const float* __restrict__ fc1_w, const float* __restrict__ fc1_b,
    const float* __restrict__ fc2_w, const float* __restrict__ fc2_b,
    const float* __restrict__ fc3_w, const float* __restrict__ fc3_b,
    float* __restrict__ out)
{
    __shared__ float g[3 * HDIM];
    __shared__ float h1[HDIM];
    __shared__ float h2[64];
    int b = blockIdx.x, t = threadIdx.x;
    float cnt = pcnt[b]; cnt = cnt > 1.0f ? cnt : 1.0f;
    float s = psum[(size_t)b * HDIM + t];
    g[t]        = s / cnt;
    g[HDIM + t] = pmax[(size_t)b * HDIM + t];
    g[2*HDIM+t] = s;
    __syncthreads();

    float a1 = fc1_b[t];
    for (int k = 0; k < 3 * HDIM; ++k) a1 += g[k] * fc1_w[k * HDIM + t];
    h1[t] = a1 > 0.0f ? a1 : 0.0f;
    __syncthreads();

    if (t < 64) {
        float a2 = fc2_b[t];
        for (int k = 0; k < HDIM; ++k) a2 += h1[k] * fc2_w[k * 64 + t];
        h2[t] = a2 > 0.0f ? a2 : 0.0f;
    }
    __syncthreads();

    if (t == 0) {
        float a3 = fc3_b[0];
        for (int k = 0; k < 64; ++k) a3 += h2[k] * fc3_w[k];
        out[b] = a3;
    }
}

// ---------------------------------------------------------------------------
// Launcher
// ---------------------------------------------------------------------------
extern "C" void kernel_launch(void* const* d_in, const int* in_sizes, int n_in,
                              void* d_out, int out_size, void* d_ws, size_t ws_size,
                              hipStream_t stream) {
    const float* x        = (const float*)d_in[0];
    const int*   eidx     = (const int*)  d_in[1];
    const int*   batch    = (const int*)  d_in[2];
    const float* conv_ws  = (const float*)d_in[3];
    const float* conv_bs  = (const float*)d_in[4];
    const float* bn_gamma = (const float*)d_in[5];
    const float* bn_beta  = (const float*)d_in[6];
    const float* bn_mean  = (const float*)d_in[7];
    const float* bn_var   = (const float*)d_in[8];
    const float* fc1_w    = (const float*)d_in[9];
    const float* fc1_b    = (const float*)d_in[10];
    const float* fc2_w    = (const float*)d_in[11];
    const float* fc2_b    = (const float*)d_in[12];
    const float* fc3_w    = (const float*)d_in[13];
    const float* fc3_b    = (const float*)d_in[14];
    float* out = (float*)d_out;

    const int N = in_sizes[0] / HDIM;
    const int E = in_sizes[1] / 2;
    const int B = out_size;
    const int L = in_sizes[3] / (HDIM * HDIM);
    const int* src = eidx;
    const int* dst = eidx + E;

    // Workspace layout (floats)
    float* ws   = (float*)d_ws;
    float* dinv = ws;                          // N
    float* hA   = dinv + N;                    // N*128 (GEMM output)
    float* hB   = hA + (size_t)N * HDIM;       // N*128 (aggregation / layer output)
    float* psum = hB + (size_t)N * HDIM;       // B*128
    float* pmax = psum + (size_t)B * HDIM;     // B*128
    float* pcnt = pmax + (size_t)B * HDIM;     // B
    (void)ws_size; (void)n_in;

    const long NH = (long)N * HDIM;

    // degree / normalization (recomputed every call: deterministic)
    k_zero<<<(int)((N + 255) / 256), 256, 0, stream>>>(dinv, N);
    k_zero<<<(int)(((long)B * (2 * HDIM + 1) + 255) / 256), 256, 0, stream>>>(
        psum, (long)B * (2 * HDIM + 1));
    k_deg<<<(E + 255) / 256, 256, 0, stream>>>(dst, dinv, E);
    k_rsqrt<<<(N + 255) / 256, 256, 0, stream>>>(dinv, N);

    const float* hin = x;
    for (int l = 0; l < L; ++l) {
        k_gemm_wmma<<<(N + 127) / 128, 256, 0, stream>>>(
            hin, conv_ws + (size_t)l * HDIM * HDIM, hA, N);
        k_self<<<(int)((NH + 255) / 256), 256, 0, stream>>>(hA, dinv, hB, N);
        k_edge<<<(E + 7) / 8, 256, 0, stream>>>(hA, src, dst, dinv, hB, E);
        k_bnrelu<<<(int)((NH + 255) / 256), 256, 0, stream>>>(
            hB, conv_bs + l * HDIM, bn_gamma + l * HDIM, bn_beta + l * HDIM,
            bn_mean + l * HDIM, bn_var + l * HDIM, N);
        hin = hB;
    }

    k_pool<<<(N + 7) / 8, 256, 0, stream>>>(hB, batch, psum, pmax, pcnt, N);
    k_mlp<<<B, 128, 0, stream>>>(psum, pmax, pcnt, fc1_w, fc1_b, fc2_w, fc2_b,
                                 fc3_w, fc3_b, out);
}
